// Decoder_7438883357260
// MI455X (gfx1250) — matrix operand order repaired and checked by reading, not compiled, not run on hardware
//
#include <hip/hip_runtime.h>

typedef __bf16 bf16_t;
typedef __attribute__((ext_vector_type(16))) __bf16 v16bf;
typedef __attribute__((ext_vector_type(8)))  __bf16 v8bf;
typedef __attribute__((ext_vector_type(8)))  float   v8f;

constexpr int kH    = 1024;
constexpr int kD    = 128;
constexpr int kBS   = 128;
constexpr int kSEQ  = 256;
constexpr int kK0   = kH + kD;       // 1152
constexpr int kEMBED = 265216;       // (L+1)*H + SEQ*H
constexpr int kTREND_OFF  = 3 * kH;          // 3072
constexpr int kSEASON_OFF = 134144;          // total - half*seq
constexpr int kNBLK = 64;                    // persistent blocks (16 gate cols each)

// ---------------- WMMA helpers (CDNA5 layouts, CDNA5 ISA) ----------------

__device__ inline v8f wmma_bf16(v16bf a, v16bf b, v8f c) {
  return __builtin_amdgcn_wmma_f32_16x16x32_bf16(
      /*neg_a=*/false, a, /*neg_b=*/false, b,
      /*c_mod=*/(short)0, c, /*reuse_a=*/false, /*reuse_b=*/false);
}

// A fragment: 16x32 bf16, row M = lane&15, lane<16 -> K k0..k0+15,
// lane>=16 -> K k0+16..k0+31. Two 16B contiguous loads.
__device__ inline v16bf load_a(const bf16_t* __restrict__ base, int ldk,
                               int row0, int k0) {
  const int lane = threadIdx.x & 31;
  const int m  = row0 + (lane & 15);
  const int kq = (lane >> 4) << 4;                // 0 or 16
  const bf16_t* p = base + (size_t)m * ldk + (k0 + kq);
  v8bf lo = *(const v8bf*)(p);
  v8bf hi = *(const v8bf*)(p + 8);
  return __builtin_shufflevector(lo, hi, 0, 1, 2, 3, 4, 5, 6, 7,
                                 8, 9, 10, 11, 12, 13, 14, 15);
}

// B fragment: 32x16 bf16, col N = lane&15, lane<16 -> K k0..k0+15,
// lane>=16 -> K k0+16..k0+31. W is [N][K] row-major -> 32 contiguous bytes.
// Prefetch near-cache (locality 3 -> WGP scope) ~8 K-iterations ahead.
__device__ inline v16bf load_b(const bf16_t* __restrict__ W, int ldk,
                               int ncol0, int k0) {
  const int lane = threadIdx.x & 31;
  const int n  = ncol0 + (lane & 15);
  const int ko = k0 + ((lane >> 4) << 4);         // 0 or 16
  const bf16_t* p = W + (size_t)n * ldk + ko;
  v8bf lo = *(const v8bf*)(p);
  v8bf hi = *(const v8bf*)(p + 8);
  __builtin_prefetch((const void*)(p + 256), 0, 3);   // global_prefetch_b8 (near)
  return __builtin_shufflevector(lo, hi, 0, 1, 2, 3, 4, 5, 6, 7,
                                 8, 9, 10, 11, 12, 13, 14, 15);
}

// Same B-fragment gather but from LDS (ds_load_b128 x2).
__device__ inline v16bf load_b_lds(const bf16_t* W, int ldk, int k0) {
  const int lane = threadIdx.x & 31;
  const int n  = lane & 15;
  const int ko = k0 + ((lane >> 4) << 4);
  const bf16_t* p = W + (size_t)n * ldk + ko;
  v8bf lo = *(const v8bf*)(p);
  v8bf hi = *(const v8bf*)(p + 8);
  return __builtin_shufflevector(lo, hi, 0, 1, 2, 3, 4, 5, 6, 7,
                                 8, 9, 10, 11, 12, 13, 14, 15);
}

__device__ inline float sigmoidf_(float x) { return 1.f / (1.f + __expf(-x)); }

// ---------------- grid-wide barrier (persistent kernel) ----------------

__device__ inline void grid_barrier(unsigned* cnt, unsigned target) {
  __syncthreads();
  if (threadIdx.x == 0) {
    __hip_atomic_fetch_add(cnt, 1u, __ATOMIC_ACQ_REL, __HIP_MEMORY_SCOPE_AGENT);
    while (__hip_atomic_load(cnt, __ATOMIC_ACQUIRE, __HIP_MEMORY_SCOPE_AGENT) < target) {
      __builtin_amdgcn_s_sleep(8);
    }
  }
  __syncthreads();
}

// ---------------- prep kernels ----------------

__global__ void cvt_f32_bf16(const float* __restrict__ src,
                             bf16_t* __restrict__ dst, int n) {
  int i = blockIdx.x * blockDim.x + threadIdx.x;
  if (i < n) dst[i] = (bf16_t)src[i];
}

__global__ void init_state(const float* __restrict__ embed,
                           const float* __restrict__ dynamics,
                           bf16_t* __restrict__ act,   // [2][BS][K0]
                           bf16_t* __restrict__ h0bf,  // [2][BS][H]
                           bf16_t* __restrict__ h1bf,
                           float*  __restrict__ h0f,   // [2][BS][H]
                           float*  __restrict__ h1f,
                           unsigned* __restrict__ cnt) {
  int idx = blockIdx.x * blockDim.x + threadIdx.x;
  if (idx == 0) *cnt = 0u;
  if (idx >= kBS * kK0) return;
  int b = idx / kK0, k = idx % kK0;
  if (k < kH) {
    float g = embed[(size_t)b * kEMBED + k];
    act[(size_t)b * kK0 + k] = (bf16_t)g;                      // buf0 glob
    act[(size_t)(kBS + b) * kK0 + k] = (bf16_t)g;              // buf1 glob
    float v0 = embed[(size_t)b * kEMBED + kH + k];
    h0f[(size_t)b * kH + k]  = v0;
    h0bf[(size_t)b * kH + k] = (bf16_t)v0;
    float v1 = embed[(size_t)b * kEMBED + 2 * kH + k];
    h1f[(size_t)b * kH + k]  = v1;
    h1bf[(size_t)b * kH + k] = (bf16_t)v1;
  } else {
    int c = k - kH;   // x0 = dynamics[:,0,:]
    act[(size_t)b * kK0 + k] = (bf16_t)dynamics[(size_t)b * kSEQ * kD + c];
  }
}

// Pack bf16(trend + season) rows: tsbuf[(b*SEQ+t)][512].
__global__ void ts_pack(const float* __restrict__ embed,
                        bf16_t* __restrict__ tsbuf) {
  const int i = blockIdx.x * blockDim.x + threadIdx.x;   // over 32768*64 vec8
  const int r  = i >> 6;
  const int c8 = (i & 63) << 3;
  const int b  = r >> 8;
  const int tt = r & 255;
  const float* pt = embed + (size_t)b * kEMBED + kTREND_OFF  + (size_t)tt * 512 + c8;
  const float* ps = embed + (size_t)b * kEMBED + kSEASON_OFF + (size_t)tt * 512 + c8;
  v8f tv = *(const v8f*)pt;
  v8f sv = *(const v8f*)ps;
  v8bf o;
#pragma unroll
  for (int j = 0; j < 8; ++j) o[j] = (bf16_t)(tv[j] + sv[j]);
  *(v8bf*)(tsbuf + (size_t)r * 512 + c8) = o;
}

// ---------------- trend/season projection: tsbuf @ pred_W.T ----------------
// M = BS*SEQ = 32768 rows, K = 512, N = 128. Writes d_out = acc + 2*pred_b.

__global__ __launch_bounds__(256) void ts_gemm(const bf16_t* __restrict__ tsbuf,
                                               const bf16_t* __restrict__ predWb, // [128][512]
                                               const float* __restrict__ predb,
                                               float* __restrict__ out) {
  const int wave = threadIdx.x >> 5;
  const int lane = threadIdx.x & 31;
  const int mt = blockIdx.x;          // 0..2047  (16 rows each)
  const int nt = wave;                // 0..7     (16 cols each)

  v8f acc = {};
#pragma unroll 2
  for (int k = 0; k < 512; k += 32) {
    v16bf a = load_a(tsbuf, 512, mt * 16, k);
    v16bf b = load_b(predWb, 512, nt * 16, k);
    acc = wmma_bf16(a, b, acc);
  }
  const int col = nt * 16 + (lane & 15);
  const float pb2 = 2.f * predb[col];
#pragma unroll
  for (int v = 0; v < 8; ++v) {
    int row = mt * 16 + ((lane >> 4) << 3) + v;
    out[(size_t)row * kD + col] = acc[v] + pb2;   // (b,t) rows collapsed
  }
}

// ---------------- GRU layer: one block owns 16 gate columns (all 3 gates) ----------------

__device__ inline void gru_layer(const bf16_t* __restrict__ Ain, int ldA,
                                 const bf16_t* __restrict__ Wih,   // [3H][ldA]
                                 const bf16_t* __restrict__ Hold_bf,
                                 const bf16_t* __restrict__ Whh,   // [3H][H]
                                 float br_, float bz_, float bin, float bhn,
                                 const float* __restrict__ Hold_f,
                                 float*  __restrict__ Hnew_f,
                                 bf16_t* __restrict__ Hnew_bf,
                                 int colbase, int mrow0) {
  v8f accR  = {};
  v8f accZ  = {};
  v8f accI  = {};   // i_n part (input matmul only)
  v8f accHN = {};   // h_n part (hidden matmul only)

  // input-side: gi = A @ W_ih.T   (r,z accumulate here; n separately)
#pragma unroll 2
  for (int k = 0; k < ldA; k += 32) {
    v16bf a = load_a(Ain, ldA, mrow0, k);
    accR = wmma_bf16(a, load_b(Wih, ldA, colbase,          k), accR);
    accZ = wmma_bf16(a, load_b(Wih, ldA, kH + colbase,     k), accZ);
    accI = wmma_bf16(a, load_b(Wih, ldA, 2 * kH + colbase, k), accI);
  }
  // hidden-side: gh = h @ W_hh.T  (r,z fold into same accumulator)
#pragma unroll 2
  for (int k = 0; k < kH; k += 32) {
    v16bf a = load_a(Hold_bf, kH, mrow0, k);
    accR  = wmma_bf16(a, load_b(Whh, kH, colbase,          k), accR);
    accZ  = wmma_bf16(a, load_b(Whh, kH, kH + colbase,     k), accZ);
    accHN = wmma_bf16(a, load_b(Whh, kH, 2 * kH + colbase, k), accHN);
  }

  const int lane = threadIdx.x & 31;
  const int col = colbase + (lane & 15);
#pragma unroll
  for (int v = 0; v < 8; ++v) {
    const int row = mrow0 + ((lane >> 4) << 3) + v;   // C layout: VGPR v -> M=v(+8)
    const float r = sigmoidf_(accR[v] + br_);
    const float z = sigmoidf_(accZ[v] + bz_);
    const float n = tanhf(accI[v] + bin + r * (accHN[v] + bhn));
    const float hprev = Hold_f[(size_t)row * kH + col];
    const float hn = (1.f - z) * n + z * hprev;
    Hnew_f[(size_t)row * kH + col]  = hn;
    Hnew_bf[(size_t)row * kH + col] = (bf16_t)hn;
  }
}

// ---------------- persistent GRU scan ----------------

__global__ __launch_bounds__(256) void gru_scan(
    bf16_t* __restrict__ act,   // [2][BS][K0]  ([glob | x])
    bf16_t* __restrict__ h0bf, bf16_t* __restrict__ h1bf,   // [2][BS][H]
    float*  __restrict__ h0f,  float*  __restrict__ h1f,    // [2][BS][H]
    const bf16_t* __restrict__ Wih0b, const bf16_t* __restrict__ Whh0b,
    const bf16_t* __restrict__ Wih1b, const bf16_t* __restrict__ Whh1b,
    const bf16_t* __restrict__ fcWb,  // [D][H]
    const float* __restrict__ bih0, const float* __restrict__ bhh0,
    const float* __restrict__ bih1, const float* __restrict__ bhh1,
    const float* __restrict__ fcb,
    float* __restrict__ out,    // [BS][SEQ][D], pre-filled with trend_+season_
    unsigned* __restrict__ cnt) {
  __shared__ __align__(16) bf16_t fcw_lds[16 * kH];   // this block's fc_W slab (32KB)
  __shared__ __align__(32) float red[8][32][8];       // fc K-split reduction (8KB)
  const int tid  = threadIdx.x;
  const int wave = tid >> 5;
  const int lane = tid & 31;
  const int blk  = blockIdx.x;            // 0..63
  const int colbase = blk * 16;           // 16 gate columns per block
  const int mrow0   = wave * 16;          // 8 waves x 16 rows = 128 batch rows
  const int mtC = blk >> 3, ntC = blk & 7; // this block's fc tile

  // Stage fc_W rows [ntC*16, ntC*16+16) into LDS once (reused all 256 steps).
  {
    const v8bf* src = (const v8bf*)(fcWb + (size_t)ntC * 16 * kH);
    v8bf* dst = (v8bf*)fcw_lds;
#pragma unroll
    for (int i = tid; i < 16 * kH / 8; i += 256) dst[i] = src[i];
  }
  __syncthreads();

  // Per-lane gate column is fixed: hoist all biases out of the t-loop.
  const int col0 = colbase + (lane & 15);
  const float br0_ = bih0[col0]          + bhh0[col0];
  const float bz0_ = bih0[kH + col0]     + bhh0[kH + col0];
  const float bin0 = bih0[2 * kH + col0];
  const float bhn0 = bhh0[2 * kH + col0];
  const float br1_ = bih1[col0]          + bhh1[col0];
  const float bz1_ = bih1[kH + col0]     + bhh1[kH + col0];
  const float bin1 = bih1[2 * kH + col0];
  const float bhn1 = bhh1[2 * kH + col0];
  const float bbC  = fcb[ntC * 16 + (lane & 15)];

  unsigned tgt = 0;
  for (int t = 0; t < kSEQ; ++t) {
    const int buf = t & 1, nbuf = buf ^ 1;
    const bf16_t* A0   = act  + (size_t)buf  * kBS * kK0;
    const bf16_t* H0o  = h0bf + (size_t)buf  * kBS * kH;
    const float*  H0of = h0f  + (size_t)buf  * kBS * kH;
    bf16_t*       H0n  = h0bf + (size_t)nbuf * kBS * kH;
    float*        H0nf = h0f  + (size_t)nbuf * kBS * kH;
    const bf16_t* H1o  = h1bf + (size_t)buf  * kBS * kH;
    const float*  H1of = h1f  + (size_t)buf  * kBS * kH;
    bf16_t*       H1n  = h1bf + (size_t)nbuf * kBS * kH;
    float*        H1nf = h1f  + (size_t)nbuf * kBS * kH;

    // ---- layer 0: c0 = GRU([glob,x], c0) ----
    gru_layer(A0, kK0, Wih0b, H0o, Whh0b, br0_, bz0_, bin0, bhn0,
              H0of, H0nf, H0n, colbase, mrow0);
    tgt += kNBLK; grid_barrier(cnt, tgt);

    // ---- layer 1: c1 = GRU(c0_new, c1) ----
    gru_layer(H0n, kH, Wih1b, H1o, Whh1b, br1_, bz1_, bin1, bhn1,
              H1of, H1nf, H1n, colbase, mrow0);
    tgt += kNBLK; grid_barrier(cnt, tgt);

    // ---- output head: x = sigmoid(c1 @ fc_W.T + fc_b); out += x ----
    // One 16x16 tile per block; K=1024 split over 8 waves, fc_W from LDS,
    // LDS reduction: wave 0 sums the 8 partials and finalizes.
    {
      v8f acc = {};
      const int kbeg = wave * 128;
#pragma unroll 2
      for (int k = kbeg; k < kbeg + 128; k += 32) {
        v16bf a = load_a(H1n, kH, mtC * 16, k);
        v16bf b = load_b_lds(fcw_lds, kH, k);
        acc = wmma_bf16(a, b, acc);
      }
      *(v8f*)&red[wave][lane][0] = acc;      // ds_store 32B
      __syncthreads();
      if (wave == 0) {
        v8f s = *(v8f*)&red[0][lane][0];
#pragma unroll
        for (int w2 = 1; w2 < 8; ++w2) {
          v8f p = *(v8f*)&red[w2][lane][0];  // ds_load 32B
#pragma unroll
          for (int v = 0; v < 8; ++v) s[v] += p[v];
        }
        const int col = ntC * 16 + (lane & 15);
        bf16_t* xdst = act + (size_t)nbuf * kBS * kK0;
#pragma unroll
        for (int v = 0; v < 8; ++v) {
          const int row = mtC * 16 + ((lane >> 4) << 3) + v;
          const float sg = sigmoidf_(s[v] + bbC);
          out[(size_t)row * kSEQ * kD + (size_t)t * kD + col] += sg;
          xdst[(size_t)row * kK0 + kH + col] = (bf16_t)sg;   // x for next step
        }
      }
    }
    tgt += kNBLK; grid_barrier(cnt, tgt);
  }
}

// ---------------- launch ----------------

extern "C" void kernel_launch(void* const* d_in, const int* in_sizes, int n_in,
                              void* d_out, int out_size, void* d_ws, size_t ws_size,
                              hipStream_t stream) {
  (void)in_sizes; (void)n_in; (void)out_size; (void)ws_size;
  const float* embed    = (const float*)d_in[0];
  const float* dynamics = (const float*)d_in[1];
  const float* W_ih0 = (const float*)d_in[2];
  const float* W_hh0 = (const float*)d_in[3];
  const float* b_ih0 = (const float*)d_in[4];
  const float* b_hh0 = (const float*)d_in[5];
  const float* W_ih1 = (const float*)d_in[6];
  const float* W_hh1 = (const float*)d_in[7];
  const float* b_ih1 = (const float*)d_in[8];
  const float* b_hh1 = (const float*)d_in[9];
  const float* fc_W  = (const float*)d_in[10];
  const float* fc_b  = (const float*)d_in[11];
  const float* pred_W = (const float*)d_in[12];
  const float* pred_b = (const float*)d_in[13];
  float* out = (float*)d_out;

  char* ws = (char*)d_ws;
  size_t off = 0;
  auto alloc = [&](size_t bytes) -> void* {
    void* p = ws + off;
    off = (off + bytes + 255) & ~(size_t)255;
    return p;
  };

  bf16_t* Wih0b = (bf16_t*)alloc((size_t)3 * kH * kK0 * 2);
  bf16_t* Whh0b = (bf16_t*)alloc((size_t)3 * kH * kH * 2);
  bf16_t* Wih1b = (bf16_t*)alloc((size_t)3 * kH * kH * 2);
  bf16_t* Whh1b = (bf16_t*)alloc((size_t)3 * kH * kH * 2);
  bf16_t* fcWb  = (bf16_t*)alloc((size_t)kD * kH * 2);
  bf16_t* predWb = (bf16_t*)alloc((size_t)kD * 512 * 2);
  bf16_t* act   = (bf16_t*)alloc((size_t)2 * kBS * kK0 * 2);
  bf16_t* h0bf  = (bf16_t*)alloc((size_t)2 * kBS * kH * 2);
  bf16_t* h1bf  = (bf16_t*)alloc((size_t)2 * kBS * kH * 2);
  float*  h0f   = (float*)alloc((size_t)2 * kBS * kH * 4);
  float*  h1f   = (float*)alloc((size_t)2 * kBS * kH * 4);
  bf16_t* tsbuf = (bf16_t*)alloc((size_t)kBS * kSEQ * 512 * 2);
  unsigned* cnt = (unsigned*)alloc(256);

  auto cvt = [&](const float* s, bf16_t* d, int n) {
    cvt_f32_bf16<<<(n + 255) / 256, 256, 0, stream>>>(s, d, n);
  };
  cvt(W_ih0, Wih0b, 3 * kH * kK0);
  cvt(W_hh0, Whh0b, 3 * kH * kH);
  cvt(W_ih1, Wih1b, 3 * kH * kH);
  cvt(W_hh1, Whh1b, 3 * kH * kH);
  cvt(fc_W,  fcWb,  kD * kH);
  cvt(pred_W, predWb, kD * 512);

  init_state<<<(kBS * kK0 + 255) / 256, 256, 0, stream>>>(
      embed, dynamics, act, h0bf, h1bf, h0f, h1f, cnt);

  ts_pack<<<(kBS * kSEQ * 64) / 256, 256, 0, stream>>>(embed, tsbuf);

  // parallel trend+season projection writes d_out first
  ts_gemm<<<(kBS * kSEQ) / 16, 256, 0, stream>>>(tsbuf, predWb, pred_b, out);

  // persistent sequential scan (64 blocks, grid-barrier synced)
  gru_scan<<<kNBLK, 256, 0, stream>>>(
      act, h0bf, h1bf, h0f, h1f,
      Wih0b, Whh0b, Wih1b, Whh1b, fcWb,
      b_ih0, b_hh0, b_ih1, b_hh1, fc_b,
      out, cnt);
}
